// FFANModel_11553462026853
// MI455X (gfx1250) — matrix operand order repaired and checked by reading, not compiled, not run on hardware
//
#include <hip/hip_runtime.h>
#include <hip/hip_bf16.h>

// ---------------------------------------------------------------------------
// FFAN model for MI455X (gfx1250, wave32, WMMA).
//   k_zero      : zero padded-embedding tensor + conv split-K accumulators
//   k_gather    : embedding gather -> padded f32 tensor ePad[b][hp][c][w]
//   k_w2prep    : w2 (900x900 f32) -> zero-padded bf16 image (912x928)
//   k_conv<KK>  : conv-as-GEMM, bf16 WMMA, split-K with f32 atomics (dominant,
//                 HBM-bound: streams 590MB of weights once, NT loads)
//   k_pool      : bias + relu + max-pool -> conv features (8x900)
//   k_qkv       : Q/K/V projections (wave-per-output dot products)
//   k_softstats : per-(b,s) softmax row max and 1/Z (rank-1 attention)
//   k_attout    : att_out[b,t] = sum_s V_s * softmax(K_s*Q_t/30)
//   k_afc       : A = [conv, att_out] @ w1^T + b1 (+ b2 folded) -> Apb
//   k_scan      : 900 sequential steps, single persistent WG, WMMA GEMM/step
//                 (kc-outer loop: B tile from LDS once per kc, 2 M-tiles/wave)
// ---------------------------------------------------------------------------

typedef __attribute__((ext_vector_type(4)))  float   v4f;
typedef __attribute__((ext_vector_type(8)))  float   v8f;
typedef __attribute__((ext_vector_type(16))) __bf16  v16bf;
typedef __attribute__((ext_vector_type(8)))  __bf16  v8bf;

#define WG_B   8
#define WG_NC  8
#define WG_CS  512
#define EMBD   64
#define HIDN   300
#define DD     900
#define DPAD   912      // 57*16
#define KPAD   928      // 29*32
#define NMT    (DPAD/16) // 57
#define YSTRIDE 320
#define SPLITK 32
#define HPAD   14       // 8 + 2*3 halo rows

__device__ __forceinline__ float wave_sum(float v){
  #pragma unroll
  for (int off = 16; off > 0; off >>= 1) v += __shfl_xor(v, off, 32);
  return v;
}
__device__ __forceinline__ float wave_max(float v){
  #pragma unroll
  for (int off = 16; off > 0; off >>= 1) v = fmaxf(v, __shfl_xor(v, off, 32));
  return v;
}

__device__ __forceinline__ v4f ldnt4(const float* p){
  return __builtin_nontemporal_load((const v4f*)p);  // streamed-once: keep L2 clean
}
__device__ __forceinline__ v4f ld4(const float* p){
  return *(const v4f*)p;
}

// pack 16 f32 -> v16bf (RNE converts)
__device__ __forceinline__ v16bf pack16(v4f f0, v4f f1, v4f f2, v4f f3){
  v16bf r;
  #pragma unroll
  for (int i = 0; i < 4; ++i){
    r[i]    = (__bf16)f0[i];
    r[4+i]  = (__bf16)f1[i];
    r[8+i]  = (__bf16)f2[i];
    r[12+i] = (__bf16)f3[i];
  }
  return r;
}

__device__ __forceinline__ v16bf cat8(v8bf lo, v8bf hi){
  return __builtin_shufflevector(lo, hi, 0,1,2,3,4,5,6,7,8,9,10,11,12,13,14,15);
}

__global__ void k_zero(float4* __restrict__ p, size_t n4){
  size_t i = (size_t)blockIdx.x*blockDim.x + threadIdx.x;
  size_t stride = (size_t)gridDim.x*blockDim.x;
  float4 z = {0.f, 0.f, 0.f, 0.f};
  for (; i < n4; i += stride) p[i] = z;
}

// ePad[b][hp][c][w], hp = h+3 for valid rows, halo rows stay zero.
__global__ void k_gather(const int* __restrict__ x, const float* __restrict__ emb,
                         float* __restrict__ ePad){
  int wid  = (blockIdx.x*blockDim.x + threadIdx.x) >> 5;   // wave per (b,h,w0)
  int lane = threadIdx.x & 31;
  if (wid >= WG_B*WG_NC*(WG_CS/32)) return;
  int w = ((wid & 15) << 5) + lane;
  int h = (wid >> 4) & 7;
  int b = wid >> 7;
  int idx = x[(b*WG_NC + h)*WG_CS + w];
  const float4* er = (const float4*)(emb + (size_t)idx*EMBD);  // 256B contiguous
  float* base = ePad + ((size_t)(b*HPAD + (h+3))*EMBD)*WG_CS + w;
  #pragma unroll
  for (int c4 = 0; c4 < 16; ++c4){
    float4 v = er[c4];
    base[(size_t)(c4*4+0)*WG_CS] = v.x;   // coalesced across lanes (w fastest)
    base[(size_t)(c4*4+1)*WG_CS] = v.y;
    base[(size_t)(c4*4+2)*WG_CS] = v.z;
    base[(size_t)(c4*4+3)*WG_CS] = v.w;
  }
}

__global__ void k_w2prep(const float* __restrict__ w2, __bf16* __restrict__ w2b){
  int i = blockIdx.x*blockDim.x + threadIdx.x;
  if (i >= DPAD*KPAD) return;
  int o = i / KPAD, j = i - o*KPAD;
  float v = (o < DD && j < DD) ? w2[o*DD + j] : 0.f;
  w2b[i] = (__bf16)v;
}

// Conv GEMM: grid (10 Mtile-pairs, SPLITK segments); 4 waves = 4 N-tiles of 16.
// A = weights (M=channels x K), B = activations (K x N=(b,h)), bf16 WMMA, f32 acc.
template<int KK>
__global__ void __launch_bounds__(128) k_conv(const float* __restrict__ W,
                                              const float* __restrict__ ePad,
                                              float* __restrict__ Y){
  const int wave = threadIdx.x >> 5;
  const int lane = threadIdx.x & 31;
  const int l16  = lane & 15;
  const bool hib = lane >= 16;
  const int s0   = hib ? 8 : 0;

  const int mt0 = blockIdx.x*2;
  const int seg = blockIdx.y;
  const int rowlen = EMBD*KK*WG_CS;
  int o0 = mt0*16 + l16;       if (o0 > HIDN-1) o0 = HIDN-1;   // clamp (dup rows ignored)
  int o1 = (mt0+1)*16 + l16;   if (o1 > HIDN-1) o1 = HIDN-1;
  const float* A0 = W + (size_t)o0*rowlen;
  const float* A1 = W + (size_t)o1*rowlen;

  const int n  = wave*16 + l16;           // output column = (b,h)
  const int bb = n >> 3, hh = n & 7;
  const int hbase = hh + (3 - (KK >> 1)); // hp = hbase + kh

  const int chunks = (EMBD*KK*WG_CS)/32;  // K/32
  const int per = chunks / SPLITK;
  const int q0 = seg*per, q1 = q0 + per;

  v8f acc0 = {0,0,0,0,0,0,0,0};
  v8f acc1 = {0,0,0,0,0,0,0,0};

  for (int q = q0; q < q1; ++q){
    const int w0 = (q & 15) << 5;         // w offset (16 chunks per (c,kh))
    const int t  = q >> 4;
    const int kh = t % KK;
    const int c  = t / KK;
    // B operand: K {s0..+7, s0+16..+23}/lane from L2-resident activation tensor
    const float* bp = ePad + (((size_t)(bb*HPAD + hbase + kh)*EMBD + c) << 9)
                           + w0 + (hib ? 8 : 0);
    v16bf Bv = pack16(ld4(bp), ld4(bp+4), ld4(bp+16), ld4(bp+20));
    // A operands: streamed weights (non-temporal), K {s0..+7, s0+16..+23}/lane
    const float* a0p = A0 + q*32 + s0;
    v16bf Av0 = pack16(ldnt4(a0p), ldnt4(a0p+4), ldnt4(a0p+16), ldnt4(a0p+20));
    const float* a1p = A1 + q*32 + s0;
    v16bf Av1 = pack16(ldnt4(a1p), ldnt4(a1p+4), ldnt4(a1p+16), ldnt4(a1p+20));
    acc0 = __builtin_amdgcn_wmma_f32_16x16x32_bf16(false, Av0, false, Bv,
                                                   (short)0, acc0, false, false);
    acc1 = __builtin_amdgcn_wmma_f32_16x16x32_bf16(false, Av1, false, Bv,
                                                   (short)0, acc1, false, false);
  }
  // C layout: vgpr r, lane -> (M = r + 8*hi, N = l16); split-K -> atomics
  float* yc0 = Y + (size_t)n*YSTRIDE + mt0*16     + (hib ? 8 : 0);
  float* yc1 = Y + (size_t)n*YSTRIDE + (mt0+1)*16 + (hib ? 8 : 0);
  #pragma unroll
  for (int r = 0; r < 8; ++r){
    atomicAdd(yc0 + r, acc0[r]);
    atomicAdd(yc1 + r, acc1[r]);
  }
}

__global__ void k_pool(const float* __restrict__ y3, const float* __restrict__ y5,
                       const float* __restrict__ y7,
                       const float* __restrict__ bc3, const float* __restrict__ bc5,
                       const float* __restrict__ bc7, float* __restrict__ convp){
  int b = blockIdx.x;
  int o = threadIdx.x;
  if (o >= HIDN) return;
  const float* ys[3] = {y3, y5, y7};
  const float* bs[3] = {bc3, bc5, bc7};
  #pragma unroll
  for (int j = 0; j < 3; ++j){
    float bias = bs[j][o];
    float m = 0.f;                        // relu floor
    #pragma unroll
    for (int h = 0; h < 8; ++h){
      float v = ys[j][(size_t)(b*8+h)*YSTRIDE + o] + bias;
      m = fmaxf(m, fmaxf(v, 0.f));
    }
    convp[b*DD + j*HIDN + o] = m;
  }
}

__global__ void k_qkv(const float* __restrict__ convp,
                      const float* __restrict__ wq, const float* __restrict__ bq,
                      const float* __restrict__ wk, const float* __restrict__ bk,
                      const float* __restrict__ wv, const float* __restrict__ bv,
                      float* __restrict__ Qm, float* __restrict__ Km,
                      float* __restrict__ Vm){
  int z = blockIdx.z;
  const float* Wm = (z==0) ? wq : (z==1) ? wk : wv;
  const float* Bm = (z==0) ? bq : (z==1) ? bk : bv;
  float*       Om = (z==0) ? Qm : (z==1) ? Km : Vm;
  int wid  = blockIdx.x*(blockDim.x>>5) + (threadIdx.x>>5);
  int lane = threadIdx.x & 31;
  if (wid >= WG_B*DD) return;
  int b = wid / DD, t = wid - b*DD;
  const float* row = Wm + (size_t)t*DD;
  const float* cv  = convp + b*DD;
  float s = 0.f;
  for (int j = lane; j < DD; j += 32) s += cv[j]*row[j];
  s = wave_sum(s);
  if (lane == 0) Om[wid] = s + Bm[t];
}

__global__ void k_softstats(const float* __restrict__ Qm, const float* __restrict__ Km,
                            float* __restrict__ rowmax, float* __restrict__ invZ){
  int wid  = blockIdx.x*(blockDim.x>>5) + (threadIdx.x>>5);
  int lane = threadIdx.x & 31;
  if (wid >= WG_B*DD) return;
  int b = wid / DD, s = wid - b*DD;
  const float inv_scale = 1.0f/30.0f;                 // 1/sqrt(900)
  float Ks = Km[b*DD + s]*inv_scale;
  const float* Qr = Qm + b*DD;
  float mx = -3.4e38f;
  for (int t = lane; t < DD; t += 32) mx = fmaxf(mx, Ks*Qr[t]);
  mx = wave_max(mx);
  float sum = 0.f;
  for (int t = lane; t < DD; t += 32) sum += __expf(Ks*Qr[t] - mx);
  sum = wave_sum(sum);
  if (lane == 0){ rowmax[wid] = mx; invZ[wid] = 1.0f/sum; }
}

__global__ void k_attout(const float* __restrict__ Qm, const float* __restrict__ Km,
                         const float* __restrict__ Vm, const float* __restrict__ rowmax,
                         const float* __restrict__ invZ, float* __restrict__ attout){
  int wid  = blockIdx.x*(blockDim.x>>5) + (threadIdx.x>>5);
  int lane = threadIdx.x & 31;
  if (wid >= WG_B*DD) return;
  int b = wid / DD, t = wid - b*DD;
  const float inv_scale = 1.0f/30.0f;
  float Qt = Qm[b*DD + t]*inv_scale;
  float s = 0.f;
  for (int ss = lane; ss < DD; ss += 32)
    s += Vm[b*DD+ss]*__expf(Km[b*DD+ss]*Qt - rowmax[b*DD+ss])*invZ[b*DD+ss];
  s = wave_sum(s);
  if (lane == 0) attout[wid] = s;
}

// Apb[b][o] = A[b,o] + b2[o] (padded rows o>=900 -> 0)
__global__ void k_afc(const float* __restrict__ convp, const float* __restrict__ attout,
                      const float* __restrict__ w1m, const float* __restrict__ b1m,
                      const float* __restrict__ b2m, float* __restrict__ Apb){
  int wid  = blockIdx.x*(blockDim.x>>5) + (threadIdx.x>>5);
  int lane = threadIdx.x & 31;
  if (wid >= WG_B*DPAD) return;
  int b = wid & 7, o = wid >> 3;
  if (o < DD){
    const float* row = w1m + (size_t)o*(2*DD);
    const float* cv  = convp + b*DD;
    const float* av  = attout + b*DD;
    float s = 0.f;
    for (int j = lane; j < DD; j += 32) s += cv[j]*row[j];
    for (int j = lane; j < DD; j += 32) s += av[j]*row[DD + j];
    s = wave_sum(s);
    if (lane == 0) Apb[b*DPAD + o] = s + b1m[o] + b2m[o];
  } else {
    if (lane == 0) Apb[b*DPAD + o] = 0.f;
  }
}

// 900 sequential steps, single persistent WG (32 waves), WMMA GEMM per step:
// S <- Apb + W2 * (S .* w_it),  w_it recomputed from rank-1 attention.
// kc is the OUTER loop of the GEMM so the LDS B tile is loaded exactly once
// per kc (no mt-invariant hoisting -> no scratch spills); each wave keeps two
// persistent accumulators for its two M-tiles (57 tiles over 32 waves).
__global__ void __launch_bounds__(1024) k_scan(const float* __restrict__ Qm,
                                               const float* __restrict__ Km,
                                               const float* __restrict__ rowmax,
                                               const float* __restrict__ invZ,
                                               const float* __restrict__ Apb,
                                               const __bf16* __restrict__ w2b,
                                               float* __restrict__ out){
  __shared__ float  Ss[WG_B][DPAD];      // state, f32          (29.2 KB)
  __shared__ __bf16 yB[16][KPAD];        // B operand image     (29.7 KB)
  const int tid  = threadIdx.x;
  const int wave = tid >> 5;
  const int lane = tid & 31;
  const int l16  = lane & 15;
  const bool hib = lane >= 16;
  const float inv_scale = 1.0f/30.0f;

  // per-wave M-tiles
  const int  mt0  = wave;                 // 0..31
  const int  mt1  = wave + 32;            // 32..63, valid if < 57
  const bool has1 = (mt1 < NMT);
  const int  o0   = mt0*16 + l16;
  const int  o1   = (has1 ? mt1 : mt0)*16 + l16;   // clamp: dup rows, discarded
  const __bf16* arow0 = w2b + (size_t)o0*KPAD;
  const __bf16* arow1 = w2b + (size_t)o1*KPAD;

  for (int i = tid; i < WG_B*DPAD; i += 1024) Ss[i/DPAD][i%DPAD] = 0.f;
  for (int i = tid; i < 8*KPAD;   i += 1024) yB[8 + i/KPAD][i%KPAD] = (__bf16)0.f;
  __syncthreads();

  for (int it = 0; it < DD; ++it){
    // build yB[b][j] = bf16(S[b][j] * attw[b,it,j])
    for (int i = tid; i < WG_B*KPAD; i += 1024){
      int b = i / KPAD, j = i - b*KPAD;
      float yv = 0.f;
      if (j < DD){
        float wv = __expf(Km[b*DD+it]*inv_scale*Qm[b*DD+j] - rowmax[b*DD+it])
                 * invZ[b*DD+it];
        yv = Ss[b][j]*wv;
      }
      yB[b][j] = (__bf16)yv;
    }
    __syncthreads();

    // GEMM: M=912 (57 tiles), N=16 (8 real batches), K=928
    v8f acc0 = {0,0,0,0,0,0,0,0};
    v8f acc1 = {0,0,0,0,0,0,0,0};
    #pragma unroll 1
    for (int kc = 0; kc < KPAD/32; ++kc){
      const __bf16* brow = &yB[l16][kc*32 + (hib ? 8 : 0)];
      v16bf Bv = cat8(*(const v8bf*)brow, *(const v8bf*)(brow + 16));
      const int as0 = kc*32 + (hib ? 8 : 0);
      v16bf Av0 = cat8(*(const v8bf*)(arow0 + as0), *(const v8bf*)(arow0 + as0 + 16));
      acc0 = __builtin_amdgcn_wmma_f32_16x16x32_bf16(false, Av0, false, Bv,
                                                     (short)0, acc0, false, false);
      v16bf Av1 = cat8(*(const v8bf*)(arow1 + as0), *(const v8bf*)(arow1 + as0 + 16));
      acc1 = __builtin_amdgcn_wmma_f32_16x16x32_bf16(false, Av1, false, Bv,
                                                     (short)0, acc1, false, false);
    }
    if (l16 < WG_B){                     // columns 0..7 are real batches
      const int b   = l16;
      const int ob0 = mt0*16 + (hib ? 8 : 0);
      #pragma unroll
      for (int r = 0; r < 8; ++r) Ss[b][ob0 + r] = Apb[b*DPAD + ob0 + r] + acc0[r];
      if (has1){
        const int ob1 = mt1*16 + (hib ? 8 : 0);
        #pragma unroll
        for (int r = 0; r < 8; ++r) Ss[b][ob1 + r] = Apb[b*DPAD + ob1 + r] + acc1[r];
      }
    }
    __syncthreads();
  }
  for (int i = tid; i < WG_B*DD; i += 1024) out[i] = Ss[i/DD][i%DD];
}

extern "C" void kernel_launch(void* const* d_in, const int* in_sizes, int n_in,
                              void* d_out, int out_size, void* d_ws, size_t ws_size,
                              hipStream_t stream) {
  const int*   x   = (const int*)d_in[0];
  const float* emb = (const float*)d_in[1];
  const float* wc3 = (const float*)d_in[2];
  const float* bc3 = (const float*)d_in[3];
  const float* wc5 = (const float*)d_in[4];
  const float* bc5 = (const float*)d_in[5];
  const float* wc7 = (const float*)d_in[6];
  const float* bc7 = (const float*)d_in[7];
  const float* wq  = (const float*)d_in[8];
  const float* bq  = (const float*)d_in[9];
  const float* wk  = (const float*)d_in[10];
  const float* bk  = (const float*)d_in[11];
  const float* wv  = (const float*)d_in[12];
  const float* bv  = (const float*)d_in[13];
  const float* w1  = (const float*)d_in[14];
  const float* b1  = (const float*)d_in[15];
  const float* w2  = (const float*)d_in[16];
  const float* b2  = (const float*)d_in[17];
  float* out = (float*)d_out;
  (void)in_sizes; (void)n_in; (void)out_size; (void)ws_size;

  char* p = (char*)d_ws;
  auto carve = [&](size_t bytes) -> char* {
    char* r = p; p += (bytes + 255) & ~(size_t)255; return r;
  };
  float* ePad   = (float*)carve(sizeof(float)*WG_B*HPAD*EMBD*WG_CS);  // 14.7 MB
  float* y3     = (float*)carve(sizeof(float)*64*YSTRIDE);
  float* y5     = (float*)carve(sizeof(float)*64*YSTRIDE);
  float* y7     = (float*)carve(sizeof(float)*64*YSTRIDE);
  char*  zero_end = p;                       // zero ePad (halo) + y accumulators
  float* convp  = (float*)carve(sizeof(float)*WG_B*DD);
  float* Qm     = (float*)carve(sizeof(float)*WG_B*DD);
  float* Km     = (float*)carve(sizeof(float)*WG_B*DD);
  float* Vm     = (float*)carve(sizeof(float)*WG_B*DD);
  float* rmx    = (float*)carve(sizeof(float)*WG_B*DD);
  float* ivZ    = (float*)carve(sizeof(float)*WG_B*DD);
  float* aout   = (float*)carve(sizeof(float)*WG_B*DD);
  float* Apb    = (float*)carve(sizeof(float)*WG_B*DPAD);
  __bf16* w2b   = (__bf16*)carve(sizeof(__bf16)*DPAD*KPAD);           // 1.69 MB

  size_t n4 = (size_t)(zero_end - (char*)d_ws) / 16;
  k_zero<<<1024, 256, 0, stream>>>((float4*)d_ws, n4);
  k_gather<<<128, 256, 0, stream>>>(x, emb, ePad);
  k_w2prep<<<(DPAD*KPAD + 255)/256, 256, 0, stream>>>(w2, w2b);

  dim3 cg(10, SPLITK, 1);
  k_conv<3><<<cg, 128, 0, stream>>>(wc3, ePad, y3);
  k_conv<5><<<cg, 128, 0, stream>>>(wc5, ePad, y5);
  k_conv<7><<<cg, 128, 0, stream>>>(wc7, ePad, y7);

  k_pool<<<8, 320, 0, stream>>>(y3, y5, y7, bc3, bc5, bc7, convp);

  dim3 qg(900, 1, 3);
  k_qkv<<<qg, 256, 0, stream>>>(convp, wq, bq, wk, bk, wv, bv, Qm, Km, Vm);
  k_softstats<<<900, 256, 0, stream>>>(Qm, Km, rmx, ivZ);
  k_attout<<<900, 256, 0, stream>>>(Qm, Km, Vm, rmx, ivZ, aout);
  k_afc<<<DPAD, 256, 0, stream>>>(convp, aout, w1, b1, b2, Apb);

  k_scan<<<1, 1024, 0, stream>>>(Qm, Km, rmx, ivZ, Apb, w2b, out);
}
